// RelationNetwork_10273561772194
// MI455X (gfx1250) — matrix operand order repaired and checked by reading, not compiled, not run on hardware
//
#include <hip/hip_runtime.h>

#define NCTX  8
#define NPAN  9
#define NPAIR 72
#define BSZ   256
#define CSZ   8
#define FSZ   512
#define HSZ   256

typedef __attribute__((ext_vector_type(16))) __bf16 v16bf;
typedef __attribute__((ext_vector_type(8)))  float  v8f;

union Frag16 { v16bf v; unsigned u[8]; };

// Pack two f32 into packed bf16 dword. Prefer hardware v_cvt_pk_bf16_f32.
__device__ __forceinline__ unsigned pack2_bf16(float a, float b) {
#if __has_builtin(__builtin_amdgcn_cvt_pk_bf16_f32)
  auto p = __builtin_amdgcn_cvt_pk_bf16_f32(a, b);
  unsigned r;
  __builtin_memcpy(&r, &p, 4);
  return r;
#else
  unsigned ua = __float_as_uint(a);
  unsigned ub = __float_as_uint(b);
  ua += 0x7FFFu + ((ua >> 16) & 1u);
  ub += 0x7FFFu + ((ub >> 16) & 1u);
  return (ua >> 16) | (ub & 0xFFFF0000u);
#endif
}

// ---------------------------------------------------------------------------
// Prep 1: panels -> bf16 row-major [4096][512] (dword-packed pairs).
// m = b*16 + q ; q<8 context panel, q>=8 choice panel.
// ---------------------------------------------------------------------------
__global__ __launch_bounds__(256) void prep_panels_kernel(
    const float* __restrict__ ctx, const float* __restrict__ cho,
    unsigned* __restrict__ panelbf) {
  const int idx = blockIdx.x * 256 + threadIdx.x;   // 0 .. 4096*256
  const int m = idx >> 8, kd = idx & 255;
  const int bb = m >> 4, q = m & 15;
  const float* src = (q < NCTX) ? (ctx + (bb * NCTX + q) * FSZ)
                                : (cho + (bb * CSZ + (q - NCTX)) * FSZ);
  panelbf[idx] = pack2_bf16(src[2 * kd], src[2 * kd + 1]);
}

// ---------------------------------------------------------------------------
// Prep 2: W1 (as Wcat=[W1a|W1b], K=512,N=512), W2, W3 -> bf16 in WMMA
// B-fragment order: dword[((ks*NT + nt)*32 + lane)*8 + v] =
//   pack(W[k][n], W[k+1][n]),  k = ks*32 + (lane>>4)*8 + (v&3)*2 + (v>>2)*16, n = nt*16+lane%16
// So each fragment is two contiguous 16B loads per lane.
// ---------------------------------------------------------------------------
__global__ __launch_bounds__(256) void prep_weights_kernel(
    const float* __restrict__ W1, const float* __restrict__ W2,
    const float* __restrict__ W3, unsigned* __restrict__ W1sw,
    unsigned* __restrict__ W2sw, unsigned* __restrict__ W3sw) {
  const int idx = blockIdx.x * 256 + threadIdx.x;   // 0 .. 196608
  if (idx < 131072) {                               // W1sw: 16 ks x 32 nt
    const int v = idx & 7, lane = (idx >> 3) & 31;
    const int nt = (idx >> 8) & 31, ks = idx >> 13;
    const int k = ks * 32 + (lane >> 4) * 8 + (v & 3) * 2 + (v >> 2) * 16;
    const int n = nt * 16 + (lane & 15);
    const float* wsrc = (n < HSZ) ? (W1 + n) : (W1 + FSZ * HSZ + (n - HSZ));
    W1sw[idx] = pack2_bf16(wsrc[k * HSZ], wsrc[(k + 1) * HSZ]);
  } else if (idx < 163840) {                        // W2sw: 8 ks x 16 nt
    const int i2 = idx - 131072;
    const int v = i2 & 7, lane = (i2 >> 3) & 31;
    const int nt = (i2 >> 8) & 15, ks = i2 >> 12;
    const int k = ks * 32 + (lane >> 4) * 8 + (v & 3) * 2 + (v >> 2) * 16;
    const int n = nt * 16 + (lane & 15);
    W2sw[i2] = pack2_bf16(W2[k * HSZ + n], W2[(k + 1) * HSZ + n]);
  } else {                                          // W3sw: 8 ks x 16 nt
    const int i3 = idx - 163840;
    const int v = i3 & 7, lane = (i3 >> 3) & 31;
    const int nt = (i3 >> 8) & 15, ks = i3 >> 12;
    const int k = ks * 32 + (lane >> 4) * 8 + (v & 3) * 2 + (v >> 2) * 16;
    const int n = nt * 16 + (lane & 15);
    W3sw[i3] = pack2_bf16(W3[k * HSZ + n], W3[(k + 1) * HSZ + n]);
  }
}

// ---------------------------------------------------------------------------
// Kernel 1: proj[m][0:256] = panel_m @ W1a + b1 ; proj[m][256:512] = @ W1b
// GEMM M=4096 K=512 N=512, bf16 WMMA, pure b128 loads (no conversion VALU).
// grid (256 m-tiles, 8 n-blocks), 128 threads = 4 waves, 1 tile/wave.
// ---------------------------------------------------------------------------
__global__ __launch_bounds__(128) void panel_proj_kernel(
    const unsigned* __restrict__ panelbf, const uint4* __restrict__ W1sw4,
    const float* __restrict__ b1, float* __restrict__ proj) {
  const int tid  = threadIdx.x;
  const int lane = tid & 31, wave = tid >> 5;
  const int lm   = lane & 15, hlf = lane >> 4;
  const int m    = blockIdx.x * 16 + lm;            // panel row
  const int ntg  = blockIdx.y * 4 + wave;           // n-tile 0..31
  const int n    = ntg * 16 + lm;
  const uint4* pan4 = (const uint4*)panelbf + m * 64;   // 256 dw/row

  v8f acc = {0.f, 0.f, 0.f, 0.f, 0.f, 0.f, 0.f, 0.f};
#pragma unroll
  for (int ks = 0; ks < FSZ / 32; ++ks) {
    Frag16 a, bf;
    const uint4 t0 = pan4[ks * 4 + hlf];
    const uint4 t1 = pan4[ks * 4 + hlf + 2];
    a.u[0] = t0.x; a.u[1] = t0.y; a.u[2] = t0.z; a.u[3] = t0.w;
    a.u[4] = t1.x; a.u[5] = t1.y; a.u[6] = t1.z; a.u[7] = t1.w;
    const int base = ((ks * 32 + ntg) * 32 + lane) * 2;
    const uint4 w0 = W1sw4[base];
    const uint4 w1 = W1sw4[base + 1];
    bf.u[0] = w0.x; bf.u[1] = w0.y; bf.u[2] = w0.z; bf.u[3] = w0.w;
    bf.u[4] = w1.x; bf.u[5] = w1.y; bf.u[6] = w1.z; bf.u[7] = w1.w;
    acc = __builtin_amdgcn_wmma_f32_16x16x32_bf16(false, a.v, false, bf.v,
                                                  (short)0, acc, false, false);
  }
  const float bias = (n < HSZ) ? b1[n] : 0.f;
#pragma unroll
  for (int r = 0; r < 8; ++r) {
    const int mo = blockIdx.x * 16 + r + 8 * hlf;
    proj[mo * (2 * HSZ) + n] = acc[r] + bias;
  }
}

// ---------------------------------------------------------------------------
// Kernel 2: one WG per (b,c).
//   h1 = relu(A_i + B_j)  (72 pairs, bf16 LDS, stride 132dw, b128-readable)
//   h2 = relu(h1 @ W2 + b2); agg = masked mean   (bf16 WMMA)
//   g  = relu(agg @ W3 + b3)                     (bf16 WMMA, M=1 padded)
//   out = g @ W4 + b4                            (tree reduce)
// ---------------------------------------------------------------------------
__global__ __launch_bounds__(256) void relation_kernel(
    const float* __restrict__ proj, const uint4* __restrict__ W2sw4,
    const uint4* __restrict__ W3sw4, const float* __restrict__ b2,
    const float* __restrict__ b3, const float* __restrict__ W4,
    const float* __restrict__ b4, float* __restrict__ out) {
  __shared__ float    Ash[NPAN * HSZ];
  __shared__ float    Bsh[NPAN * HSZ];
  __shared__ __align__(16) unsigned h1sh[80 * 132];  // 80x256 bf16, stride 132 dw
  __shared__ float    agg_sh[HSZ];
  __shared__ unsigned aggbf[HSZ / 2];
  __shared__ float    g_sh[HSZ];
  __shared__ float    red_sh[HSZ];

  const int tid = threadIdx.x;
  const int bc  = blockIdx.x;
  const int bb  = bc >> 3, cc = bc & 7;

  // Stage panel projections (f32) for this (b,c).
  for (int idx = tid; idx < NPAN * HSZ; idx += 256) {
    const int q = idx >> 8, k = idx & 255;
    const int row = (q < NCTX) ? (bb * 16 + q) : (bb * 16 + NCTX + cc);
    Ash[idx] = proj[row * (2 * HSZ) + k];
    Bsh[idx] = proj[row * (2 * HSZ) + HSZ + k];
  }
  __syncthreads();

  // Build h1 bf16 (rows 72..79 zero).
  for (int idx = tid; idx < 80 * 128; idx += 256) {
    const int p = idx >> 7, kk = idx & 127;
    unsigned pk = 0u;
    if (p < NPAIR) {
      const int i = p >> 3, jr = p & 7;
      const int j = jr + (jr >= i ? 1 : 0);
      float v0 = Ash[i * 256 + 2 * kk]     + Bsh[j * 256 + 2 * kk];
      float v1 = Ash[i * 256 + 2 * kk + 1] + Bsh[j * 256 + 2 * kk + 1];
      v0 = v0 > 0.f ? v0 : 0.f;
      v1 = v1 > 0.f ? v1 : 0.f;
      pk = pack2_bf16(v0, v1);
    }
    h1sh[p * 132 + kk] = pk;
  }
  __syncthreads();

  // Layer-2 GEMM: [80x256] x [256x256]; wave owns cols [32w,32w+32).
  const int lane = tid & 31, wave = tid >> 5;
  const int lm   = lane & 15, hlf = lane >> 4;
  const int n0 = wave * 32 + lm, n1 = n0 + 16;
  const uint4* h4 = (const uint4*)h1sh;             // 33 uint4 per row

  v8f acc[5][2];
#pragma unroll
  for (int mt = 0; mt < 5; ++mt)
#pragma unroll
    for (int nt = 0; nt < 2; ++nt)
#pragma unroll
      for (int r = 0; r < 8; ++r) acc[mt][nt][r] = 0.f;

#pragma unroll
  for (int ks = 0; ks < 8; ++ks) {
    Frag16 bf0, bf1;
    {
      const int base0 = ((ks * 16 + wave * 2) * 32 + lane) * 2;
      const uint4 w0 = W2sw4[base0],     w1 = W2sw4[base0 + 1];
      const uint4 w2 = W2sw4[base0 + 64], w3 = W2sw4[base0 + 65]; // nt+1 -> +32*2
      bf0.u[0] = w0.x; bf0.u[1] = w0.y; bf0.u[2] = w0.z; bf0.u[3] = w0.w;
      bf0.u[4] = w1.x; bf0.u[5] = w1.y; bf0.u[6] = w1.z; bf0.u[7] = w1.w;
      bf1.u[0] = w2.x; bf1.u[1] = w2.y; bf1.u[2] = w2.z; bf1.u[3] = w2.w;
      bf1.u[4] = w3.x; bf1.u[5] = w3.y; bf1.u[6] = w3.z; bf1.u[7] = w3.w;
    }
#pragma unroll
    for (int mt = 0; mt < 5; ++mt) {
      Frag16 a;
      const int rb = (mt * 16 + lm) * 33 + ks * 4 + hlf;
      const uint4 t0 = h4[rb];
      const uint4 t1 = h4[rb + 2];
      a.u[0] = t0.x; a.u[1] = t0.y; a.u[2] = t0.z; a.u[3] = t0.w;
      a.u[4] = t1.x; a.u[5] = t1.y; a.u[6] = t1.z; a.u[7] = t1.w;
      acc[mt][0] = __builtin_amdgcn_wmma_f32_16x16x32_bf16(
          false, a.v, false, bf0.v, (short)0, acc[mt][0], false, false);
      acc[mt][1] = __builtin_amdgcn_wmma_f32_16x16x32_bf16(
          false, a.v, false, bf1.v, (short)0, acc[mt][1], false, false);
    }
  }

  // relu(h2 + b2), masked mean over 72 valid rows.
  const float c0 = b2[n0], c1 = b2[n1];
  float part0 = 0.f, part1 = 0.f;
#pragma unroll
  for (int mt = 0; mt < 5; ++mt)
#pragma unroll
    for (int r = 0; r < 8; ++r) {
      const int p = mt * 16 + r + 8 * hlf;
      if (p < NPAIR) {
        const float x0 = acc[mt][0][r] + c0;
        const float x1 = acc[mt][1][r] + c1;
        part0 += x0 > 0.f ? x0 : 0.f;
        part1 += x1 > 0.f ? x1 : 0.f;
      }
    }
  part0 += __shfl_xor(part0, 16, 32);
  part1 += __shfl_xor(part1, 16, 32);
  if (hlf == 0) {
    agg_sh[n0] = part0 * (1.f / (float)NPAIR);
    agg_sh[n1] = part1 * (1.f / (float)NPAIR);
  }
  __syncthreads();

  // Layer 3 via WMMA: A = agg padded to 16 rows (rows 1..15 zero).
  if (tid < 128) aggbf[tid] = pack2_bf16(agg_sh[2 * tid], agg_sh[2 * tid + 1]);
  __syncthreads();

  v8f acc3[2];
#pragma unroll
  for (int nt = 0; nt < 2; ++nt)
#pragma unroll
    for (int r = 0; r < 8; ++r) acc3[nt][r] = 0.f;

#pragma unroll
  for (int ks = 0; ks < 8; ++ks) {
    Frag16 a, bf0, bf1;
#pragma unroll
    for (int r = 0; r < 8; ++r) {
      const unsigned v = aggbf[ks * 16 + (r & 3) + ((r >> 2) << 3) + hlf * 4];
      a.u[r] = (lm == 0) ? v : 0u;
    }
    const int base0 = ((ks * 16 + wave * 2) * 32 + lane) * 2;
    const uint4 w0 = W3sw4[base0],      w1 = W3sw4[base0 + 1];
    const uint4 w2 = W3sw4[base0 + 64], w3 = W3sw4[base0 + 65];
    bf0.u[0] = w0.x; bf0.u[1] = w0.y; bf0.u[2] = w0.z; bf0.u[3] = w0.w;
    bf0.u[4] = w1.x; bf0.u[5] = w1.y; bf0.u[6] = w1.z; bf0.u[7] = w1.w;
    bf1.u[0] = w2.x; bf1.u[1] = w2.y; bf1.u[2] = w2.z; bf1.u[3] = w2.w;
    bf1.u[4] = w3.x; bf1.u[5] = w3.y; bf1.u[6] = w3.z; bf1.u[7] = w3.w;
    acc3[0] = __builtin_amdgcn_wmma_f32_16x16x32_bf16(
        false, a.v, false, bf0.v, (short)0, acc3[0], false, false);
    acc3[1] = __builtin_amdgcn_wmma_f32_16x16x32_bf16(
        false, a.v, false, bf1.v, (short)0, acc3[1], false, false);
  }
  if (hlf == 0) {                       // D row M=0 lives in VGPR0, lanes 0-15
    float g0 = acc3[0][0] + b3[n0];
    float g1 = acc3[1][0] + b3[n1];
    g_sh[n0] = g0 > 0.f ? g0 : 0.f;
    g_sh[n1] = g1 > 0.f ? g1 : 0.f;
  }
  __syncthreads();

  // Layer 4: score = g . W4 + b4
  red_sh[tid] = g_sh[tid] * W4[tid];
  __syncthreads();
#pragma unroll
  for (int s = 128; s > 0; s >>= 1) {
    if (tid < s) red_sh[tid] += red_sh[tid + s];
    __syncthreads();
  }
  if (tid == 0) out[bc] = red_sh[0] + b4[0];
}

// ---------------------------------------------------------------------------
extern "C" void kernel_launch(void* const* d_in, const int* in_sizes, int n_in,
                              void* d_out, int out_size, void* d_ws, size_t ws_size,
                              hipStream_t stream) {
  const float* ctx = (const float*)d_in[0];
  const float* cho = (const float*)d_in[1];
  const float* W1  = (const float*)d_in[2];
  const float* b1  = (const float*)d_in[3];
  const float* W2  = (const float*)d_in[4];
  const float* b2  = (const float*)d_in[5];
  const float* W3  = (const float*)d_in[6];
  const float* b3  = (const float*)d_in[7];
  const float* W4  = (const float*)d_in[8];
  const float* b4  = (const float*)d_in[9];
  float* out = (float*)d_out;

  // Workspace layout (12.75 MB):
  char* ws = (char*)d_ws;
  float*    proj    = (float*)ws;                                   // 8 MB
  unsigned* panelbf = (unsigned*)(ws + (8u << 20));                 // 4 MB
  unsigned* W1sw    = (unsigned*)(ws + (12u << 20));                // 512 KB
  unsigned* W2sw    = (unsigned*)(ws + (12u << 20) + (512u << 10)); // 128 KB
  unsigned* W3sw    = (unsigned*)(ws + (12u << 20) + (640u << 10)); // 128 KB

  prep_panels_kernel<<<dim3(4096), 256, 0, stream>>>(ctx, cho, panelbf);
  prep_weights_kernel<<<dim3(768), 256, 0, stream>>>(W1, W2, W3, W1sw, W2sw, W3sw);
  panel_proj_kernel<<<dim3(256, 8), 128, 0, stream>>>(
      panelbf, (const uint4*)W1sw, b1, proj);
  relation_kernel<<<dim3(BSZ * CSZ), 256, 0, stream>>>(
      proj, (const uint4*)W2sw, (const uint4*)W3sw, b2, b3, W4, b4, out);
}
